// GCN_24034636989227
// MI455X (gfx1250) — matrix operand order repaired and checked by reading, not compiled, not run on hardware
//
#include <hip/hip_runtime.h>
#include <hip/hip_bf16.h>

// ---------------------------------------------------------------------------
// 2-layer GAT (single head) for MI455X / gfx1250, wave32.
//  - GEMMs: v_wmma_f32_16x16x32_bf16, with LDS tiles stored in *fragment
//    order* so each fragment is two ds_load_b128 per lane (no scalar u16s).
//  - Edge softmax + aggregation: streaming kernels, float4 loads, f32 atomics.
// ---------------------------------------------------------------------------

#define NODES   100000
#define NEDGE   1600000
#define ETOT    (NEDGE + NODES)   // edges + self loops
#define IN_F    256
#define HID_F   128
#define OUT_F   64
#define NEG_SLOPE 0.2f

typedef __attribute__((ext_vector_type(16))) __bf16 v16bf;
typedef __attribute__((ext_vector_type(8)))  float  v8f;

union FragBF { v16bf v; uint4 q[2]; };

__device__ __forceinline__ unsigned short f32_to_bf16(float f) {
    unsigned u = __float_as_uint(f);
    u += 0x7fffu + ((u >> 16) & 1u);      // round-to-nearest-even
    return (unsigned short)(u >> 16);
}

__device__ __forceinline__ void edge_sd(const int* __restrict__ ei, int e,
                                        int& s, int& d) {
    if (e < NEDGE) { s = ei[e]; d = ei[NEDGE + e]; }
    else           { s = e - NEDGE; d = s; }       // appended self-loop
}

// Fragment-order index helpers (ISA 7.12.2 VGPR layouts, wave32).
// A (16x32 bf16 tile): lane = (row&15) + 16*hsel(k); element = (k&7)+8*(k>>4)
__device__ __forceinline__ int a_frag_idx(int r, int k) {
    int lane = (r & 15) + 16 * ((k >> 3) & 1);
    int e    = (k & 7) + 8 * (k >> 4);
    return ((r >> 4) * 32 + lane) * 16 + e;   // [rowTile][lane][e] ushorts
}
// B (32x16 bf16 tile): lane = (col&15) + 16*hsel(k); element = (k&7)+8*(k>>4)
__device__ __forceinline__ int b_frag_idx(int k, int col) {
    int lane = (col & 15) + 16 * ((k >> 3) & 1);
    int e    = (k & 7) + 8 * (k >> 4);
    return ((col >> 4) * 32 + lane) * 16 + e; // [colTile][lane][e] ushorts
}

// ---------------------------------------------------------------------------
// GEMM: H[nrows,FD] = X[nrows,KD] @ W[KD,FD], bf16 WMMA, f32 accumulate.
// Block = 256 threads (8 waves), 128 rows per block, full FD width.
// Wave w computes rows [w*16, w*16+16) x FD via FD/16 WMMA accumulators.
// ---------------------------------------------------------------------------
template <int KD, int FD>
__global__ __launch_bounds__(256)
void gemm_wmma_bf16(const float* __restrict__ X, const float* __restrict__ W,
                    float* __restrict__ H, int nrows) {
    constexpr int NT = FD / 16;          // 16x16 column tiles per wave
    // fragment-order tiles: [tile][lane 0..31][16 bf16] -> 2x uint4 per lane
    __shared__ uint4 sAq[8  * 32 * 2];   // 128x32 A tile   (8 KB)
    __shared__ uint4 sBq[NT * 32 * 2];   // 32xFD  W tile   (<= 8 KB)
    unsigned short* sA = (unsigned short*)sAq;
    unsigned short* sB = (unsigned short*)sBq;

    const int tid  = threadIdx.x;
    const int lane = tid & 31;
    const int wv   = tid >> 5;           // wave id 0..7
    const int lrow = lane & 15;
    const int hi   = lane >> 4;          // 0: lanes 0-15, 1: lanes 16-31
    const int rowBase = blockIdx.x * 128;

    v8f acc[NT];
    #pragma unroll
    for (int c = 0; c < NT; ++c)
        #pragma unroll
        for (int i = 0; i < 8; ++i) acc[c][i] = 0.0f;

    for (int k0 = 0; k0 < KD; k0 += 32) {
        __syncthreads();
        // stage A: 128 rows x 32 K, f32 -> bf16, permuted to fragment order
        for (int i = tid; i < 128 * 32; i += 256) {
            int r = i >> 5, c = i & 31;
            int gr = rowBase + r;
            float v = (gr < nrows) ? X[(size_t)gr * KD + (k0 + c)] : 0.0f;
            sA[a_frag_idx(r, c)] = f32_to_bf16(v);
        }
        // stage W: 32 K x FD, f32 -> bf16, permuted to fragment order
        for (int i = tid; i < 32 * FD; i += 256) {
            int k = i / FD, col = i % FD;
            sB[b_frag_idx(k, col)] = f32_to_bf16(W[(size_t)(k0 + k) * FD + col]);
        }
        __syncthreads();

        FragBF a;                                     // 2x ds_load_b128
        a.q[0] = sAq[(wv * 32 + lane) * 2 + 0];
        a.q[1] = sAq[(wv * 32 + lane) * 2 + 1];

        #pragma unroll
        for (int c = 0; c < NT; ++c) {
            FragBF b;                                 // 2x ds_load_b128
            b.q[0] = sBq[(c * 32 + lane) * 2 + 0];
            b.q[1] = sBq[(c * 32 + lane) * 2 + 1];
            acc[c] = __builtin_amdgcn_wmma_f32_16x16x32_bf16(
                         false, a.v, false, b.v, (short)0, acc[c],
                         false, false);
        }
    }

    // C/D layout: lanes 0-15 -> N=lane, M=vgpr; lanes 16-31 -> M=8+vgpr
    #pragma unroll
    for (int c = 0; c < NT; ++c) {
        int col = c * 16 + lrow;
        #pragma unroll
        for (int v = 0; v < 8; ++v) {
            int row = rowBase + wv * 16 + hi * 8 + v;
            if (row < nrows) H[(size_t)row * FD + col] = acc[c][v];
        }
    }
}

// ---------------------------------------------------------------------------
// alpha_s[n] = H[n,:]·a_src, alpha_d[n] = H[n,:]·a_dst  (one wave per node)
// float4 loads: F=128 -> all 32 lanes, F=64 -> lanes 0..15
// ---------------------------------------------------------------------------
template <int F>
__global__ __launch_bounds__(256)
void alpha_kernel(const float* __restrict__ H, const float* __restrict__ a_src,
                  const float* __restrict__ a_dst, float* __restrict__ as,
                  float* __restrict__ ad, int n) {
    int wid  = (int)((blockIdx.x * (size_t)blockDim.x + threadIdx.x) >> 5);
    int lane = threadIdx.x & 31;
    if (wid >= n) return;
    float ps = 0.0f, pd = 0.0f;
    if (lane < F / 4) {
        float4 h = *(const float4*)&H[(size_t)wid * F + lane * 4];
        float4 s = *(const float4*)&a_src[lane * 4];
        float4 d = *(const float4*)&a_dst[lane * 4];
        ps = h.x * s.x + h.y * s.y + h.z * s.z + h.w * s.w;
        pd = h.x * d.x + h.y * d.y + h.z * d.z + h.w * d.w;
    }
    #pragma unroll
    for (int m = 16; m; m >>= 1) {
        ps += __shfl_xor(ps, m, 32);
        pd += __shfl_xor(pd, m, 32);
    }
    if (lane == 0) { as[wid] = ps; ad[wid] = pd; }
}

// order-preserving float<->uint mapping for atomicMax-based segment max
__device__ __forceinline__ unsigned f2key(float f) {
    unsigned b = __float_as_uint(f);
    return (b & 0x80000000u) ? ~b : (b | 0x80000000u);
}
__device__ __forceinline__ float key2f(unsigned k) {
    unsigned b = (k & 0x80000000u) ? (k & 0x7fffffffu) : ~k;
    return __uint_as_float(b);
}

__global__ __launch_bounds__(256)
void edge_max_kernel(const float* __restrict__ as, const float* __restrict__ ad,
                     const int* __restrict__ ei, unsigned* __restrict__ mkeys) {
    int e = blockIdx.x * blockDim.x + threadIdx.x;
    if (e >= ETOT) return;
    int s, d; edge_sd(ei, e, s, d);
    float v = as[s] + ad[d];
    v = (v > 0.0f) ? v : NEG_SLOPE * v;            // leaky_relu
    atomicMax(&mkeys[d], f2key(v));                // global_atomic_max_u32
}

__global__ __launch_bounds__(256)
void edge_exp_kernel(const float* __restrict__ as, const float* __restrict__ ad,
                     const int* __restrict__ ei, const unsigned* __restrict__ mkeys,
                     float* __restrict__ ex, float* __restrict__ denom) {
    int e = blockIdx.x * blockDim.x + threadIdx.x;
    if (e >= ETOT) return;
    int s, d; edge_sd(ei, e, s, d);
    float v = as[s] + ad[d];
    v = (v > 0.0f) ? v : NEG_SLOPE * v;
    float t = __expf(v - key2f(mkeys[d]));
    ex[e] = t;
    atomicAdd(&denom[d], t);                       // global_atomic_add_f32
}

// out[dst,:] += H[src,:] * ex[e]/denom[dst]   (one wave per edge, grid-stride)
// float4 loads; F=128 -> 32 lanes, F=64 -> lanes 0..15
template <int F>
__global__ __launch_bounds__(256)
void aggregate_kernel(const float* __restrict__ H, const int* __restrict__ ei,
                      const float* __restrict__ ex, const float* __restrict__ denom,
                      float* __restrict__ out) {
    int wid  = (int)((blockIdx.x * (size_t)blockDim.x + threadIdx.x) >> 5);
    int lane = threadIdx.x & 31;
    int nw   = (int)((gridDim.x * (size_t)blockDim.x) >> 5);
    for (int e = wid; e < ETOT; e += nw) {
        int s, d; edge_sd(ei, e, s, d);
        float coef = ex[e] / denom[d];
        if (lane < F / 4) {
            float4 h = *(const float4*)&H[(size_t)s * F + lane * 4];
            float* o = &out[(size_t)d * F + lane * 4];
            atomicAdd(o + 0, h.x * coef);
            atomicAdd(o + 1, h.y * coef);
            atomicAdd(o + 2, h.z * coef);
            atomicAdd(o + 3, h.w * coef);
        }
    }
}

__global__ void fill_f32(float* p, float v, size_t n) {
    size_t i = blockIdx.x * (size_t)blockDim.x + threadIdx.x;
    if (i < n) p[i] = v;
}
__global__ void fill_u32(unsigned* p, unsigned v, size_t n) {
    size_t i = blockIdx.x * (size_t)blockDim.x + threadIdx.x;
    if (i < n) p[i] = v;
}
// in-place x = relu(x + b[f])
__global__ void relu_bias_kernel(float* x, const float* __restrict__ b,
                                 size_t n, int Fmask) {
    size_t i = blockIdx.x * (size_t)blockDim.x + threadIdx.x;
    if (i < n) {
        float v = x[i] + b[i & (size_t)Fmask];
        x[i] = v > 0.0f ? v : 0.0f;
    }
}
// in-place x += b[f]
__global__ void bias_kernel(float* x, const float* __restrict__ b,
                            size_t n, int Fmask) {
    size_t i = blockIdx.x * (size_t)blockDim.x + threadIdx.x;
    if (i < n) x[i] += b[i & (size_t)Fmask];
}

// ---------------------------------------------------------------------------
static inline int cdiv(long long a, long long b) { return (int)((a + b - 1) / b); }

extern "C" void kernel_launch(void* const* d_in, const int* in_sizes, int n_in,
                              void* d_out, int out_size, void* d_ws, size_t ws_size,
                              hipStream_t stream) {
    const float* x     = (const float*)d_in[0];
    const int*   ei    = (const int*)  d_in[1];
    const float* W1    = (const float*)d_in[2];
    const float* a1s   = (const float*)d_in[3];
    const float* a1d   = (const float*)d_in[4];
    const float* b1    = (const float*)d_in[5];
    const float* W2    = (const float*)d_in[6];
    const float* a2s   = (const float*)d_in[7];
    const float* a2d   = (const float*)d_in[8];
    const float* b2    = (const float*)d_in[9];
    float* out = (float*)d_out;

    // workspace carve-up (floats); H buffer reused for layer-2 features
    float* ws = (float*)d_ws;
    size_t o = 0;
    float*    H     = ws + o; o += (size_t)NODES * HID_F;  // H1, later H2
    float*    AGG   = ws + o; o += (size_t)NODES * HID_F;  // agg1 -> X2 (in place)
    float*    as_   = ws + o; o += NODES;
    float*    ad_   = ws + o; o += NODES;
    unsigned* mkeys = (unsigned*)(ws + o); o += NODES;
    float*    denom = ws + o; o += NODES;
    float*    ex    = ws + o; o += ETOT;
    (void)ws_size; (void)n_in; (void)in_sizes; (void)out_size;

    const int edgeBlocks = cdiv(ETOT, 256);
    const int nodeWaveBlocks = cdiv((long long)NODES * 32, 256);
    const int aggBlocks = 4096;

    // ---------------- layer 1 ----------------
    fill_f32<<<cdiv((size_t)NODES * HID_F, 256), 256, 0, stream>>>(AGG, 0.0f, (size_t)NODES * HID_F);
    fill_u32<<<cdiv(NODES, 256), 256, 0, stream>>>(mkeys, 0u, NODES);
    fill_f32<<<cdiv(NODES, 256), 256, 0, stream>>>(denom, 0.0f, NODES);

    gemm_wmma_bf16<IN_F, HID_F><<<cdiv(NODES, 128), 256, 0, stream>>>(x, W1, H, NODES);
    alpha_kernel<HID_F><<<nodeWaveBlocks, 256, 0, stream>>>(H, a1s, a1d, as_, ad_, NODES);
    edge_max_kernel<<<edgeBlocks, 256, 0, stream>>>(as_, ad_, ei, mkeys);
    edge_exp_kernel<<<edgeBlocks, 256, 0, stream>>>(as_, ad_, ei, mkeys, ex, denom);
    aggregate_kernel<HID_F><<<aggBlocks, 256, 0, stream>>>(H, ei, ex, denom, AGG);
    relu_bias_kernel<<<cdiv((size_t)NODES * HID_F, 256), 256, 0, stream>>>(AGG, b1, (size_t)NODES * HID_F, HID_F - 1);

    // ---------------- layer 2 ----------------
    fill_u32<<<cdiv(NODES, 256), 256, 0, stream>>>(mkeys, 0u, NODES);
    fill_f32<<<cdiv(NODES, 256), 256, 0, stream>>>(denom, 0.0f, NODES);
    fill_f32<<<cdiv((size_t)NODES * OUT_F, 256), 256, 0, stream>>>(out, 0.0f, (size_t)NODES * OUT_F);

    gemm_wmma_bf16<HID_F, OUT_F><<<cdiv(NODES, 128), 256, 0, stream>>>(AGG, W2, H, NODES);
    alpha_kernel<OUT_F><<<nodeWaveBlocks, 256, 0, stream>>>(H, a2s, a2d, as_, ad_, NODES);
    edge_max_kernel<<<edgeBlocks, 256, 0, stream>>>(as_, ad_, ei, mkeys);
    edge_exp_kernel<<<edgeBlocks, 256, 0, stream>>>(as_, ad_, ei, mkeys, ex, denom);
    aggregate_kernel<OUT_F><<<aggBlocks, 256, 0, stream>>>(H, ei, ex, denom, out);
    bias_kernel<<<cdiv((size_t)NODES * OUT_F, 256), 256, 0, stream>>>(out, b2, (size_t)NODES * OUT_F, OUT_F - 1);
}
